// InteractionDecoder_82429012345237
// MI455X (gfx1250) — matrix operand order repaired and checked by reading, not compiled, not run on hardware
//
#include <hip/hip_runtime.h>

// ---------------------------------------------------------------------------
// MI455X (gfx1250) implementation.
//   * bf16 WMMA (v_wmma_f32_16x16x32_bf16) for layer-1 (128x128) AND layer-2
//     (128x{3,3,1,3} zero-padded to 128x16) of every MLP.
//   * node MLP hoisted from per-edge (2*E rows) to per-node (N rows).
//   * Weights staged once per persistent block into LDS (CDNA5: 320KB/WGP).
// ---------------------------------------------------------------------------

#define E_EDGES   500000
#define NN_NODES  100000
#define L         128
#define TILES_E   (E_EDGES/16)   // 31250 (exact)
#define TILES_N   (NN_NODES/16)  //  6250 (exact)

#define W1P 136   // bf16 pitch for transposed W1/W2pad rows in LDS (bank pad)
#define HP  132   // float pitch for per-wave hidden[16][128] (16B-aligned rows)

typedef __attribute__((ext_vector_type(16))) __bf16 v16bf;
typedef __attribute__((ext_vector_type(8)))  __bf16 v8bf;
typedef __attribute__((ext_vector_type(8)))  float  v8f;

// ---------------------------------------------------------------------------
// Kernel 1: convert + transpose W1 matrices (f32 [k][c] -> bf16 [c][k]) once.
// Order in workspace: i1, i2, f_scaler, comp_corr (edge MLPs), node_weight.
// ---------------------------------------------------------------------------
__global__ void prep_w1(const float* __restrict__ s0, const float* __restrict__ s1,
                        const float* __restrict__ s2, const float* __restrict__ s3,
                        const float* __restrict__ s4, __bf16* __restrict__ dst) {
    const float* src[5] = {s0, s1, s2, s3, s4};
    int i = blockIdx.x * blockDim.x + threadIdx.x;
    const int total = 5 * L * L;
    for (; i < total; i += gridDim.x * blockDim.x) {
        const int m = i / (L * L);
        const int r = (i % (L * L)) / L;   // output row = column c of W1
        const int k = i % L;
        dst[m * L * L + r * L + k] = (__bf16)src[m][k * L + r];
    }
}

// ---------------------------------------------------------------------------
// A operand (16-bit, 16x32 per K-step) per ISA 7.12.2:
//   lane L: row m = L&15; half = L>>4
//   elems 0..7  -> K = s*32 + half*8 + {0..7}
//   elems 8..15 -> K = s*32 + 16 + half*8 + {0..7}
// ---------------------------------------------------------------------------
__device__ __forceinline__ v16bf pack_a(float4 p0, float4 p1, float4 p2, float4 p3) {
    v16bf t;
    t[0]  = (__bf16)p0.x; t[1]  = (__bf16)p0.y; t[2]  = (__bf16)p0.z; t[3]  = (__bf16)p0.w;
    t[4]  = (__bf16)p1.x; t[5]  = (__bf16)p1.y; t[6]  = (__bf16)p1.z; t[7]  = (__bf16)p1.w;
    t[8]  = (__bf16)p2.x; t[9]  = (__bf16)p2.y; t[10] = (__bf16)p2.z; t[11] = (__bf16)p2.w;
    t[12] = (__bf16)p3.x; t[13] = (__bf16)p3.y; t[14] = (__bf16)p3.z; t[15] = (__bf16)p3.w;
    return t;
}

__device__ __forceinline__ void load_a_global(const float* __restrict__ xrow,
                                              int lane, v16bf a[4]) {
    const int half = lane >> 4;
#pragma unroll
    for (int s = 0; s < 4; ++s) {
        const int k0 = s * 32 + half * 8;
        const int k1 = k0 + 16;
        float4 p0 = *(const float4*)(xrow + k0);
        float4 p1 = *(const float4*)(xrow + k0 + 4);
        float4 p2 = *(const float4*)(xrow + k1);
        float4 p3 = *(const float4*)(xrow + k1 + 4);
        a[s] = pack_a(p0, p1, p2, p3);
    }
}

// B operand (32x16 per K-step): lane L holds column c, K = s*32 + (L>>4)*8
// + {0..7} and s*32 + 16 + (L>>4)*8 + {0..7} (matches A) -> 2x ds_load_b128.
__device__ __forceinline__ v16bf load_b(const __bf16* __restrict__ wt,
                                        int c, int s, int half) {
    const __bf16* bp = wt + c * W1P + s * 32 + half * 8;   // 16B aligned
    v8bf lo = *(const v8bf*)bp;
    v8bf hi = *(const v8bf*)(bp + 16);
    return __builtin_shufflevector(lo, hi, 0, 1, 2, 3, 4, 5, 6, 7,
                                   8, 9, 10, 11, 12, 13, 14, 15);
}

// ---------------------------------------------------------------------------
// hidden = relu(X @ W1 + b1) for one 16-row tile; f32 result into per-wave LDS.
// All 4 B K-step operands preloaded so the 8 ds_load_b128 issue as one clause
// before 4 back-to-back chained WMMAs.
// D layout: VGPR r = row (L>>4)*8+r, lane = column.
// ---------------------------------------------------------------------------
__device__ __forceinline__ void mlp_hidden(const v16bf a[4],
                                           const __bf16* __restrict__ lw1t,
                                           const float* __restrict__ lb1,
                                           float* __restrict__ hid, int lane) {
    const int cb   = lane & 15;
    const int half = lane >> 4;
#pragma unroll
    for (int n = 0; n < 8; ++n) {
        const int c = n * 16 + cb;
        v16bf b[4];
#pragma unroll
        for (int s = 0; s < 4; ++s) b[s] = load_b(lw1t, c, s, half);
        v8f acc = {0.f, 0.f, 0.f, 0.f, 0.f, 0.f, 0.f, 0.f};
#pragma unroll
        for (int s = 0; s < 4; ++s)
            acc = __builtin_amdgcn_wmma_f32_16x16x32_bf16(
                false, a[s], false, b[s], (short)0, acc, false, false);
        const float bias = lb1[c];
#pragma unroll
        for (int r = 0; r < 8; ++r) {
            float h = acc[r] + bias;
            h = h > 0.f ? h : 0.f;
            hid[(half * 8 + r) * HP + c] = h;
        }
    }
}

// ---------------------------------------------------------------------------
// Layer 2 via WMMA: out16 = hidden[16,128] @ W2pad[128,16] (cols>=outc are 0).
// Hidden is read back from LDS in A layout (f32 -> bf16). D has lane=col, so
// cols 0..2 are scattered through a tiny per-wave LDS buffer back to lane=row.
// ---------------------------------------------------------------------------
__device__ __forceinline__ void mlp_out_wmma(const float* __restrict__ hid,
                                             const __bf16* __restrict__ lw2b,
                                             const float* __restrict__ lb2,
                                             float* __restrict__ outb,
                                             int lane, float res[3]) {
    const int row  = lane & 15;
    const int half = lane >> 4;
    v16bf a2[4], b2[4];
#pragma unroll
    for (int s = 0; s < 4; ++s) {
        const int k0 = s * 32 + half * 8;
        const int k1 = k0 + 16;
        float4 h0 = *(const float4*)(hid + row * HP + k0);
        float4 h1 = *(const float4*)(hid + row * HP + k0 + 4);
        float4 h2 = *(const float4*)(hid + row * HP + k1);
        float4 h3 = *(const float4*)(hid + row * HP + k1 + 4);
        a2[s] = pack_a(h0, h1, h2, h3);
        b2[s] = load_b(lw2b, row, s, half);   // row == lane&15 == B column
    }
    v8f o = {0.f, 0.f, 0.f, 0.f, 0.f, 0.f, 0.f, 0.f};
#pragma unroll
    for (int s = 0; s < 4; ++s)
        o = __builtin_amdgcn_wmma_f32_16x16x32_bf16(
            false, a2[s], false, b2[s], (short)0, o, false, false);
    if (row < 3) {                       // lanes 0..2 / 16..18 hold real cols
#pragma unroll
        for (int r = 0; r < 8; ++r) outb[(half * 8 + r) * 4 + row] = o[r];
    }
    // same-wave LDS ordering guarantees stores above complete before this load
    float4 v = *(const float4*)(outb + row * 4);
    res[0] = v.x + lb2[0];
    res[1] = v.y + lb2[1];
    res[2] = v.z + lb2[2];
}

// ---------------------------------------------------------------------------
// Kernel 2: wnode[n] = MLP(node_latent[n]) once per node (not per edge-end).
// ---------------------------------------------------------------------------
struct NodeArgs {
    const float*  nlat;
    const __bf16* w1t;   // transposed bf16 W1 (node_weight)
    const float*  b1;
    const float*  w2;    // [128][1]
    const float*  b2;    // [1]
    float*        wnode; // [N]
};

__launch_bounds__(256, 1)
__global__ void node_kernel(NodeArgs A) {
    extern __shared__ char smem[];
    __bf16* lw1t    = (__bf16*)smem;                    // 128*W1P
    __bf16* lw2b    = lw1t + L * W1P;                   // 16*W1P (padded W2)
    float*  lb1     = (float*)(smem + (L + 16) * W1P * sizeof(__bf16));
    float*  lb2     = lb1 + L;                          // 16
    float*  hidAll  = lb2 + 16;                         // 8 waves * 16*HP
    float*  outbAll = hidAll + 8 * 16 * HP;             // 8 waves * 64

    const int tid = threadIdx.x;
    for (int i = tid; i < L * L; i += blockDim.x) {
        const int c = i >> 7, k = i & 127;
        lw1t[c * W1P + k] = A.w1t[c * L + k];
    }
    for (int i = tid; i < 16 * L; i += blockDim.x) {
        const int c = i >> 7, k = i & 127;
        lw2b[c * W1P + k] = (c == 0) ? (__bf16)A.w2[k] : (__bf16)0.f;
    }
    for (int i = tid; i < L; i += blockDim.x) lb1[i] = A.b1[i];
    if (tid < 16) lb2[tid] = (tid == 0) ? A.b2[0] : 0.f;
    __syncthreads();

    const int lane   = tid & 31;
    const int wave   = tid >> 5;
    float*    hid    = hidAll + wave * 16 * HP;
    float*    outb   = outbAll + wave * 64;
    const int nwaves = gridDim.x * (blockDim.x >> 5);
    const int wgid   = blockIdx.x * (blockDim.x >> 5) + wave;

    for (int tile = wgid; tile < TILES_N; tile += nwaves) {
        if (tile + nwaves < TILES_N)
            __builtin_prefetch(A.nlat + (size_t)(tile + nwaves) * 16 * L + (lane & 15) * L, 0, 1);
        v16bf a[4];
        const float* xrow = A.nlat + ((size_t)tile * 16 + (lane & 15)) * L;
        load_a_global(xrow, lane, a);
        mlp_hidden(a, lw1t, lb1, hid, lane);
        float r[3];
        mlp_out_wmma(hid, lw2b, lb2, outb, lane, r);
        if (lane < 16) A.wnode[(size_t)tile * 16 + lane] = r[0];
    }
}

// ---------------------------------------------------------------------------
// Kernel 3: all four edge MLPs + per-edge geometry, fused.
// ---------------------------------------------------------------------------
struct EdgeArgs {
    const int*    eidx;                   // [2][E]
    const float*  spos;                   // [E][3]
    const float*  rpos;                   // [E][3]
    const float*  va;  const float* vb;  const float* vc;  // [E][3]
    const float*  xlat;                   // [E][128]
    const __bf16* w1t;                    // 4 transposed bf16 W1s, contiguous
    const float*  b1[4];
    const float*  w2[4];                  // [128][outc]
    const float*  b2[4];
    const float*  wnode;                  // [N] precomputed
    float*        out;                    // fij | tau | dx, each [E][3]
};

__launch_bounds__(256, 1)
__global__ void edge_kernel(EdgeArgs A) {
    extern __shared__ char smem[];
    __bf16* lw1t    = (__bf16*)smem;                     // 4*128*W1P
    __bf16* lw2b    = lw1t + 4 * L * W1P;                // 4*16*W1P (padded W2)
    float*  lb1     = (float*)(smem + (4 * L + 4 * 16) * W1P * sizeof(__bf16));
    float*  lb2     = lb1 + 4 * L;                       // 16
    float*  hidAll  = lb2 + 16;                          // 8 waves * 16*HP
    float*  outbAll = hidAll + 8 * 16 * HP;              // 8 waves * 64

    const int tid = threadIdx.x;
    for (int i = tid; i < 4 * L * L; i += blockDim.x) {
        const int m = i >> 14;
        const int rc = i & 16383;
        const int c = rc >> 7, k = rc & 127;
        lw1t[m * L * W1P + c * W1P + k] = A.w1t[(size_t)m * L * L + c * L + k];
    }
    for (int i = tid; i < 4 * 16 * L; i += blockDim.x) {
        const int m = i >> 11;
        const int rc = i & 2047;
        const int c = rc >> 7, k = rc & 127;
        const int oc = (m == 2) ? 1 : 3;
        const float v = (c < oc) ? A.w2[m][k * oc + c] : 0.f;
        lw2b[m * 16 * W1P + c * W1P + k] = (__bf16)v;
    }
    for (int i = tid; i < 4 * L; i += blockDim.x) lb1[i] = A.b1[i >> 7][i & 127];
    if (tid < 16) {
        const int m = tid >> 2, c = tid & 3;
        const int oc = (m == 2) ? 1 : 3;
        lb2[tid] = (c < oc) ? A.b2[m][c] : 0.f;
    }
    __syncthreads();

    const int lane   = tid & 31;
    const int wave   = tid >> 5;
    float*    hid    = hidAll + wave * 16 * HP;
    float*    outb   = outbAll + wave * 64;
    const int nwaves = gridDim.x * (blockDim.x >> 5);
    const int wgid   = blockIdx.x * (blockDim.x >> 5) + wave;

    for (int tile = wgid; tile < TILES_E; tile += nwaves) {
        if (tile + nwaves < TILES_E)
            __builtin_prefetch(A.xlat + (size_t)(tile + nwaves) * 16 * L + (lane & 15) * L, 0, 1);

        v16bf a[4];
        const float* xrow = A.xlat + ((size_t)tile * 16 + (lane & 15)) * L;
        load_a_global(xrow, lane, a);

        float cf[3], ca[3], lamv[3], cc[3];
        mlp_hidden(a, lw1t,               lb1,         hid, lane);
        mlp_out_wmma(hid, lw2b,                lb2,      outb, lane, cf);
        mlp_hidden(a, lw1t + 1 * L * W1P, lb1 + L,     hid, lane);
        mlp_out_wmma(hid, lw2b + 1 * 16 * W1P, lb2 + 4,  outb, lane, ca);
        mlp_hidden(a, lw1t + 2 * L * W1P, lb1 + 2 * L, hid, lane);
        mlp_out_wmma(hid, lw2b + 2 * 16 * W1P, lb2 + 8,  outb, lane, lamv);
        mlp_hidden(a, lw1t + 3 * L * W1P, lb1 + 3 * L, hid, lane);
        mlp_out_wmma(hid, lw2b + 3 * 16 * W1P, lb2 + 12, outb, lane, cc);

        if (lane < 16) {
            const size_t e  = (size_t)tile * 16 + lane;
            const int    si = A.eidx[e];
            const int    ri = A.eidx[E_EDGES + e];
            const float  ws = A.wnode[si];
            const float  wr = A.wnode[ri];

            const float sx = A.spos[e * 3 + 0], sy = A.spos[e * 3 + 1], sz = A.spos[e * 3 + 2];
            const float rx = A.rpos[e * 3 + 0], ry = A.rpos[e * 3 + 1], rz = A.rpos[e * 3 + 2];
            const float ax = A.va[e * 3 + 0],   ay = A.va[e * 3 + 1],   az = A.va[e * 3 + 2];
            const float bx = A.vb[e * 3 + 0],   by = A.vb[e * 3 + 1],   bz = A.vb[e * 3 + 2];
            const float cx = A.vc[e * 3 + 0],   cy = A.vc[e * 3 + 1],   cz = A.vc[e * 3 + 2];

            const float fx = cf[0] * ax + cf[1] * bx + cf[2] * cx;
            const float fy = cf[0] * ay + cf[1] * by + cf[2] * cy;
            const float fz = cf[0] * az + cf[1] * bz + cf[2] * cz;
            const float qx = ca[0] * ax + ca[1] * bx + ca[2] * cx;
            const float qy = ca[0] * ay + ca[1] * by + ca[2] * cy;
            const float qz = ca[0] * az + ca[1] * bz + ca[2] * cz;
            const float dx = cc[0] * ax + cc[1] * bx + cc[2] * cx;
            const float dy = cc[0] * ay + cc[1] * by + cc[2] * cy;
            const float dz = cc[0] * az + cc[1] * bz + cc[2] * cz;

            const float lam = lamv[0];
            const float inv = 1.f / (ws + wr);
            const float lx  = rx - (ws * sx + wr * rx) * inv;
            const float ly  = ry - (ws * sy + wr * ry) * inv;
            const float lz  = rz - (ws * sz + wr * rz) * inv;
            const float gx = fx * lam, gy = fy * lam, gz = fz * lam;
            const float tx = ly * gz - lz * gy;
            const float ty = lz * gx - lx * gz;
            const float tz = lx * gy - ly * gx;

            float* of = A.out + e * 3;
            float* ot = A.out + (size_t)3 * E_EDGES + e * 3;
            float* od = A.out + (size_t)6 * E_EDGES + e * 3;
            of[0] = fx;      of[1] = fy;      of[2] = fz;
            ot[0] = qx - tx; ot[1] = qy - ty; ot[2] = qz - tz;
            od[0] = dx;      od[1] = dy;      od[2] = dz;
        }
    }
}

// ---------------------------------------------------------------------------
// Host launcher. Input order (setup_inputs insertion order, params expanded
// as W1,b1,W2,b2 per MLP in dict order i1,i2,f_scaler,node_weight,comp_corr):
//  0 edge_index  1 spos  2 rpos  3 va  4 vb  5 vc  6 interaction_latent
//  7 node_latent
//  8..11  i1{W1,b1,W2,b2}   12..15 i2{...}   16..19 f_scaler{...}
// 20..23 node_weight{...}   24..27 comp_corr{...}
// ---------------------------------------------------------------------------
extern "C" void kernel_launch(void* const* d_in, const int* in_sizes, int n_in,
                              void* d_out, int out_size, void* d_ws, size_t ws_size,
                              hipStream_t stream) {
    const int*   eidx = (const int*)d_in[0];
    const float* spos = (const float*)d_in[1];
    const float* rpos = (const float*)d_in[2];
    const float* va   = (const float*)d_in[3];
    const float* vb   = (const float*)d_in[4];
    const float* vc   = (const float*)d_in[5];
    const float* xlat = (const float*)d_in[6];
    const float* nlat = (const float*)d_in[7];

    // Workspace layout: wnode f32[N] at 0; five bf16 W1T matrices after.
    float*  wnode = (float*)d_ws;
    __bf16* w1t   = (__bf16*)((char*)d_ws + 400384);   // 100000*4 rounded up

    // W1 sources in workspace order: i1, i2, f_scaler, comp_corr, node_weight.
    prep_w1<<<160, 256, 0, stream>>>((const float*)d_in[8], (const float*)d_in[12],
                                     (const float*)d_in[16], (const float*)d_in[24],
                                     (const float*)d_in[20], w1t);

    NodeArgs na;
    na.nlat  = nlat;
    na.w1t   = w1t + (size_t)4 * L * L;
    na.b1    = (const float*)d_in[21];
    na.w2    = (const float*)d_in[22];
    na.b2    = (const float*)d_in[23];
    na.wnode = wnode;
    const size_t ldsN = (size_t)(L + 16) * W1P * sizeof(__bf16) +
                        (size_t)(L + 16 + 8 * 16 * HP + 8 * 64) * sizeof(float);
    node_kernel<<<128, 256, ldsN, stream>>>(na);

    EdgeArgs ea;
    ea.eidx = eidx; ea.spos = spos; ea.rpos = rpos;
    ea.va = va; ea.vb = vb; ea.vc = vc;
    ea.xlat = xlat;
    ea.w1t  = w1t;
    ea.b1[0] = (const float*)d_in[9];  ea.b1[1] = (const float*)d_in[13];
    ea.b1[2] = (const float*)d_in[17]; ea.b1[3] = (const float*)d_in[25];
    ea.w2[0] = (const float*)d_in[10]; ea.w2[1] = (const float*)d_in[14];
    ea.w2[2] = (const float*)d_in[18]; ea.w2[3] = (const float*)d_in[26];
    ea.b2[0] = (const float*)d_in[11]; ea.b2[1] = (const float*)d_in[15];
    ea.b2[2] = (const float*)d_in[19]; ea.b2[3] = (const float*)d_in[27];
    ea.wnode = wnode;
    ea.out   = (float*)d_out;
    const size_t ldsE = (size_t)(4 * L + 4 * 16) * W1P * sizeof(__bf16) +
                        (size_t)(4 * L + 16 + 8 * 16 * HP + 8 * 64) * sizeof(float);
    edge_kernel<<<512, 256, ldsE, stream>>>(ea);
}
